// GNN_Energy_Model_2nd_51874615001423
// MI455X (gfx1250) — matrix operand order repaired and checked by reading, not compiled, not run on hardware
//
#include <hip/hip_runtime.h>

#define DEV __device__ __forceinline__

typedef __attribute__((ext_vector_type(16))) _Float16 v16h;
typedef __attribute__((ext_vector_type(8)))  float    v8f;

constexpr int Bc = 8, Tc = 4096, Mc = 32768, Dc = 128;
constexpr int NROWN = Bc * Tc * 2;      // 65536 node rows
constexpr int NROWE = Bc * Mc * 4;      // 1048576 edge rows

constexpr int AST = 136;   // padded LDS stride (halfs) for K<=128 tiles
constexpr int HST = 264;   // padded LDS stride (halfs) for K=256 tile

// ---------------- WMMA fragment helpers (layouts per CDNA5 ISA 7.12.2) ----

// A-matrix 16x32 f16 fragment from an LDS tile (row-major, padded stride).
// lane 0-15: M=lane, K {0..7,16..23}; lane 16-31: M=lane-16, K {8..15,24..31}
// K pairs are memory-contiguous -> compiler emits 2x ds_load_b128 per frag.
DEV v16h load_a_lds(const _Float16* tile, int stride, int kbase) {
  const int lane = threadIdx.x & 31;
  const _Float16* p = tile + (lane & 15) * stride;
  const int hb = (lane & 16) ? 8 : 0;
  v16h a;
#pragma unroll
  for (int v = 0; v < 8; ++v) {
    int k = kbase + ((v & 4) << 2) + ((v & 3) << 1) + hb;
    a[2 * v]     = p[k];
    a[2 * v + 1] = p[k + 1];
  }
  return a;
}

// B fragment from pre-packed f16 weights: one aligned 32-byte load per lane.
// Packed layout: [kt][nt][lane][16 halfs]
DEV v16h load_b_packed(const _Float16* __restrict__ P, int ntiles, int kt, int nt) {
  const int lane = threadIdx.x & 31;
  const v16h* p = (const v16h*)(P + (((size_t)(kt * ntiles + nt) * 32 + lane) << 4));
  return *p;
}

DEV v8f wmma_f16(v16h a, v16h b, v8f c) {
  return __builtin_amdgcn_wmma_f32_16x16x32_f16(false, a, false, b,
                                                (short)0, c, false, false);
}

// ---------------- Kernels --------------------------------------------------

// Re-pack a row-major f32 weight W[Din x Dout] into WMMA B-fragment order f16.
// dst[((kt*ntiles + nt)*32 + lane)*16 + h] = W[kt*32 + (lane>>4)*8 + (h&7) + (h&8)*2][nt*16 + (lane&15)]
__global__ void k_pack_w(_Float16* __restrict__ dst, const float* __restrict__ W,
                         int Din, int Dout) {
  int idx = blockIdx.x * 256 + threadIdx.x;
  if (idx >= Din * Dout) return;
  const int ntiles = Dout >> 4;
  int h    = idx & 15;
  int lane = (idx >> 4) & 31;
  int frag = idx >> 9;
  int nt = frag % ntiles;
  int kt = frag / ntiles;
  int K = kt * 32 + ((lane >> 4) << 3) + (h & 7) + ((h & 8) << 1);
  int N = nt * 16 + (lane & 15);
  dst[idx] = (_Float16)W[(size_t)K * Dout + N];
}

__global__ void k_init(float* __restrict__ acc, const float* __restrict__ xn,
                       int relu, int n) {
  int i = blockIdx.x * blockDim.x + threadIdx.x;
  if (i < n) {
    float v = xn[i];
    if (relu) v = fmaxf(v, 0.f);
    acc[i] = v;
  }
}

// one block per (b,m); 256 threads = 2 channels x 128 features
__global__ void k_scatter(float* __restrict__ acc, const float* __restrict__ xn,
                          const float* __restrict__ A,
                          const long long* __restrict__ e0,
                          const long long* __restrict__ e1, int relu) {
  const int blk = blockIdx.x;
  const int b = blk >> 15;          // / Mc
  const int m = blk & (Mc - 1);
  const int t = threadIdx.x;
  const int c = t >> 7, d = t & 127;
  const long long src = e1[m], dst = e0[m];
  float v = xn[(((size_t)b * Tc + (size_t)src) * 2 + c) * Dc + d];
  if (relu) v = fmaxf(v, 0.f);
  v *= A[(size_t)b * Mc + m];
  float* p = &acc[(((size_t)b * Tc + (size_t)dst) * 2 + c) * Dc + d];
  unsafeAtomicAdd(p, v);            // global_atomic_add_f32
}

// out[row,:] = in[row,:] @ W + bias ; 32 rows/WG, 8 waves x 16 cols, 2 row tiles
__global__ void k_node_gemm(float* __restrict__ out, const float* __restrict__ in,
                            const _Float16* __restrict__ Wp,
                            const float* __restrict__ bias) {
  __shared__ _Float16 As[32 * AST];
  const int tid = threadIdx.x;
  const size_t row0 = (size_t)blockIdx.x * 32;
  for (int e = tid; e < 32 * 128; e += 256) {
    int r = e >> 7, k = e & 127;
    As[r * AST + k] = (_Float16)in[(row0 + r) * Dc + k];
  }
  __syncthreads();
  const int wave = tid >> 5, lane = tid & 31;
  v8f c0 = {}, c1 = {};
#pragma unroll
  for (int kt = 0; kt < 4; ++kt) {
    v16h b = load_b_packed(Wp, 8, kt, wave);
    c0 = wmma_f16(load_a_lds(As,            AST, kt * 32), b, c0);
    c1 = wmma_f16(load_a_lds(As + 16 * AST, AST, kt * 32), b, c1);
  }
  const int n = wave * 16 + (lane & 15);
  const int mb = (lane & 16) ? 8 : 0;
  const float bn = bias[n];
#pragma unroll
  for (int r = 0; r < 8; ++r) {
    out[(row0 + mb + r) * Dc + n]      = c0[r] + bn;
    out[(row0 + 16 + mb + r) * Dc + n] = c1[r] + bn;
  }
}

// xe_out = act( (xe_in + ((ei+ej) @ Wm + bm)) @ We + be ) ; 32 rows/WG
__global__ void k_edge(float* __restrict__ xe_out, const float* __restrict__ xe_in,
                       const float* __restrict__ xn,
                       const long long* __restrict__ e0,
                       const long long* __restrict__ e1,
                       const _Float16* __restrict__ Wmp, const float* __restrict__ bm,
                       const _Float16* __restrict__ Wep, const float* __restrict__ be,
                       int relu) {
  __shared__ _Float16 Ss[32 * AST];
  __shared__ _Float16 Us[32 * AST];
  __shared__ size_t baseI[32], baseJ[32];
  const int tid = threadIdx.x;
  const size_t g0 = (size_t)blockIdx.x * 32;
  if (tid < 32) {
    size_t g = g0 + tid;
    int b   = (int)(g >> 17);                 // / (Mc*4)
    int rem = (int)(g & ((size_t)(Mc * 4) - 1));
    int m = rem >> 2, ch = rem & 3;           // ni = ch>>1, nj = ch&1
    long long i0 = e0[m], i1 = e1[m];
    baseI[tid] = (((size_t)b * Tc + (size_t)i0) * 2 + (ch >> 1)) * Dc;
    baseJ[tid] = (((size_t)b * Tc + (size_t)i1) * 2 + (ch & 1)) * Dc;
  }
  __syncthreads();
  for (int e = tid; e < 32 * 128; e += 256) {
    int r = e >> 7, k = e & 127;
    Ss[r * AST + k] = (_Float16)(xn[baseI[r] + k] + xn[baseJ[r] + k]);
  }
  __syncthreads();
  const int wave = tid >> 5, lane = tid & 31;
  const int n  = wave * 16 + (lane & 15);
  const int mb = (lane & 16) ? 8 : 0;
  // GEMM1: u = s @ Wm + bm + xe_in
  v8f c0 = {}, c1 = {};
#pragma unroll
  for (int kt = 0; kt < 4; ++kt) {
    v16h b = load_b_packed(Wmp, 8, kt, wave);
    c0 = wmma_f16(load_a_lds(Ss,            AST, kt * 32), b, c0);
    c1 = wmma_f16(load_a_lds(Ss + 16 * AST, AST, kt * 32), b, c1);
  }
  const float bn = bm[n];
#pragma unroll
  for (int r = 0; r < 8; ++r) {
    Us[(mb + r) * AST + n] =
        (_Float16)(c0[r] + bn + xe_in[(g0 + mb + r) * Dc + n]);
    Us[(16 + mb + r) * AST + n] =
        (_Float16)(c1[r] + bn + xe_in[(g0 + 16 + mb + r) * Dc + n]);
  }
  __syncthreads();
  // GEMM2: v = u @ We + be
  v8f d0 = {}, d1 = {};
#pragma unroll
  for (int kt = 0; kt < 4; ++kt) {
    v16h b = load_b_packed(Wep, 8, kt, wave);
    d0 = wmma_f16(load_a_lds(Us,            AST, kt * 32), b, d0);
    d1 = wmma_f16(load_a_lds(Us + 16 * AST, AST, kt * 32), b, d1);
  }
  const float bn2 = be[n];
#pragma unroll
  for (int r = 0; r < 8; ++r) {
    float v0 = d0[r] + bn2;
    float v1 = d1[r] + bn2;
    if (relu) { v0 = fmaxf(v0, 0.f); v1 = fmaxf(v1, 0.f); }
    xe_out[(g0 + mb + r) * Dc + n]      = v0;
    xe_out[(g0 + 16 + mb + r) * Dc + n] = v1;
  }
}

// fused 128 -> 256(relu) -> 128(relu) -> 1 readout MLP ; 32 rows/WG
__global__ void k_readout(float* __restrict__ out, const float* __restrict__ X,
                          const _Float16* __restrict__ W1p, const float* __restrict__ b1,
                          const _Float16* __restrict__ W2p, const float* __restrict__ b2,
                          const float* __restrict__ W3, const float* __restrict__ b3) {
  __shared__ _Float16 Xs[32 * AST];
  __shared__ _Float16 H1[32 * HST];
  __shared__ _Float16 H2[32 * AST];
  const int tid = threadIdx.x;
  const size_t row0 = (size_t)blockIdx.x * 32;
  for (int e = tid; e < 32 * 128; e += 256) {
    int r = e >> 7, k = e & 127;
    Xs[r * AST + k] = (_Float16)X[(row0 + r) * Dc + k];
  }
  __syncthreads();
  const int wave = tid >> 5, lane = tid & 31;
  const int mb = (lane & 16) ? 8 : 0;
  // layer 1: 128 -> 256, relu ; each wave: 2 col tiles x 2 row tiles
  {
    v8f c00 = {}, c01 = {}, c10 = {}, c11 = {};
#pragma unroll
    for (int kt = 0; kt < 4; ++kt) {
      v16h a0 = load_a_lds(Xs,            AST, kt * 32);
      v16h a1 = load_a_lds(Xs + 16 * AST, AST, kt * 32);
      v16h bq0 = load_b_packed(W1p, 16, kt, wave * 2);
      v16h bq1 = load_b_packed(W1p, 16, kt, wave * 2 + 1);
      c00 = wmma_f16(a0, bq0, c00);
      c10 = wmma_f16(a1, bq0, c10);
      c01 = wmma_f16(a0, bq1, c01);
      c11 = wmma_f16(a1, bq1, c11);
    }
#pragma unroll
    for (int tcol = 0; tcol < 2; ++tcol) {
      const int n = (wave * 2 + tcol) * 16 + (lane & 15);
      const float bn = b1[n];
      const v8f& r0 = tcol ? c01 : c00;
      const v8f& r1 = tcol ? c11 : c10;
#pragma unroll
      for (int r = 0; r < 8; ++r) {
        H1[(mb + r) * HST + n]      = (_Float16)fmaxf(r0[r] + bn, 0.f);
        H1[(16 + mb + r) * HST + n] = (_Float16)fmaxf(r1[r] + bn, 0.f);
      }
    }
  }
  __syncthreads();
  // layer 2: 256 -> 128, relu
  {
    v8f d0 = {}, d1 = {};
#pragma unroll
    for (int kt = 0; kt < 8; ++kt) {
      v16h b = load_b_packed(W2p, 8, kt, wave);
      d0 = wmma_f16(load_a_lds(H1,            HST, kt * 32), b, d0);
      d1 = wmma_f16(load_a_lds(H1 + 16 * HST, HST, kt * 32), b, d1);
    }
    const int n = wave * 16 + (lane & 15);
    const float bn = b2[n];
#pragma unroll
    for (int r = 0; r < 8; ++r) {
      H2[(mb + r) * AST + n]      = (_Float16)fmaxf(d0[r] + bn, 0.f);
      H2[(16 + mb + r) * AST + n] = (_Float16)fmaxf(d1[r] + bn, 0.f);
    }
  }
  __syncthreads();
  // layer 3: 128 -> 1
  if (tid < 32) {
    float s = b3[0];
#pragma unroll 8
    for (int k = 0; k < 128; ++k)
      s += (float)H2[tid * AST + k] * W3[k];
    out[row0 + tid] = s;
  }
}

// ---------------- Launch ---------------------------------------------------

extern "C" void kernel_launch(void* const* d_in, const int* in_sizes, int n_in,
                              void* d_out, int out_size, void* d_ws, size_t ws_size,
                              hipStream_t stream) {
  const float*     x1      = (const float*)d_in[0];       // (B,T,2,D)
  const float*     x2      = (const float*)d_in[1];       // (B,M,4,D)
  const float*     A       = (const float*)d_in[2];       // (B,1,M)
  const long long* edge    = (const long long*)d_in[3];   // (2,M) int64
  const float*     node_W  = (const float*)d_in[4];
  const float*     node_b  = (const float*)d_in[5];
  const float*     mlp_W   = (const float*)d_in[6];
  const float*     mlp_b   = (const float*)d_in[7];
  const float*     edgeL_W = (const float*)d_in[8];
  const float*     edgeL_b = (const float*)d_in[9];
  const float*     nro_W1  = (const float*)d_in[10];
  const float*     nro_b1  = (const float*)d_in[11];
  const float*     nro_W2  = (const float*)d_in[12];
  const float*     nro_b2  = (const float*)d_in[13];
  const float*     nro_W3  = (const float*)d_in[14];
  const float*     nro_b3  = (const float*)d_in[15];
  const float*     ero_W1  = (const float*)d_in[16];
  const float*     ero_b1  = (const float*)d_in[17];
  const float*     ero_W2  = (const float*)d_in[18];
  const float*     ero_b2  = (const float*)d_in[19];
  const float*     ero_W3  = (const float*)d_in[20];
  const float*     ero_b3  = (const float*)d_in[21];

  // ---- workspace layout: packed f16 weights, then f32 tensors ----
  constexpr int WSQ = Dc * Dc;          // 16384 halfs per DxD matrix
  constexpr int WSR = Dc * 2 * Dc;      // 32768 halfs per 128x256 matrix
  _Float16* nodeWp  = (_Float16*)d_ws;            // 3 * WSQ
  _Float16* mlpWp   = nodeWp  + 3 * WSQ;
  _Float16* edgeLWp = mlpWp   + 3 * WSQ;
  _Float16* nro1p   = edgeLWp + 3 * WSQ;          // WSR
  _Float16* nro2p   = nro1p   + WSR;
  _Float16* ero1p   = nro2p   + WSR;
  _Float16* ero2p   = ero1p   + WSR;
  float* acc = (float*)(ero2p + WSR);
  float* xn  = acc + (size_t)NROWN * Dc;          // pre-relu node features
  float* xe  = xn  + (size_t)NROWN * Dc;          // edge features

  const long long* e0 = edge;
  const long long* e1 = edge + Mc;

  // ---- pack all weight matrices to f16 fragment layout ----
  const int gSq = (Dc * Dc + 255) / 256, gRe = (2 * Dc * Dc + 255) / 256;
  for (int i = 0; i < 3; ++i) {
    k_pack_w<<<gSq, 256, 0, stream>>>(nodeWp  + i * WSQ, node_W  + i * WSQ, Dc, Dc);
    k_pack_w<<<gSq, 256, 0, stream>>>(mlpWp   + i * WSQ, mlp_W   + i * WSQ, Dc, Dc);
    k_pack_w<<<gSq, 256, 0, stream>>>(edgeLWp + i * WSQ, edgeL_W + i * WSQ, Dc, Dc);
  }
  k_pack_w<<<gRe, 256, 0, stream>>>(nro1p, nro_W1, Dc, 2 * Dc);
  k_pack_w<<<gRe, 256, 0, stream>>>(nro2p, nro_W2, 2 * Dc, Dc);
  k_pack_w<<<gRe, 256, 0, stream>>>(ero1p, ero_W1, Dc, 2 * Dc);
  k_pack_w<<<gRe, 256, 0, stream>>>(ero2p, ero_W2, 2 * Dc, Dc);

  // ---- 3 GNN layers ----
  for (int i = 0; i < 3; ++i) {
    const int reluIn = (i > 0);                   // xn buffer holds pre-relu
    const float* xn_in = (i == 0) ? x1 : xn;
    k_init<<<(NROWN * Dc + 255) / 256, 256, 0, stream>>>(acc, xn_in, reluIn,
                                                         NROWN * Dc);
    k_scatter<<<Bc * Mc, 256, 0, stream>>>(acc, xn_in, A, e0, e1, reluIn);
    k_node_gemm<<<NROWN / 32, 256, 0, stream>>>(xn, acc, nodeWp + i * WSQ,
                                                node_b + i * Dc);
    const float* xe_in = (i == 0) ? x2 : xe;
    k_edge<<<NROWE / 32, 256, 0, stream>>>(xe, xe_in, xn, e0, e1,
                                           mlpWp + i * WSQ, mlp_b + i * Dc,
                                           edgeLWp + i * WSQ, edgeL_b + i * Dc,
                                           (i < 2) ? 1 : 0);
  }

  // ---- readouts ----
  float* out = (float*)d_out;
  k_readout<<<NROWN / 32, 256, 0, stream>>>(out, xn, nro1p, nro_b1, nro2p,
                                            nro_b2, nro_W3, nro_b3);
  k_readout<<<NROWE / 32, 256, 0, stream>>>(out + NROWN, xe, ero1p, ero_b1,
                                            ero2p, ero_b2, ero_W3, ero_b3);
}
